// AgentMatchingDecoder_70265664962758
// MI455X (gfx1250) — matrix operand order repaired and checked by reading, not compiled, not run on hardware
//
#include <hip/hip_runtime.h>
#include <hip/hip_bf16.h>

// ---------------------------------------------------------------------------
// MI455X / gfx1250 AgentMatchingDecoder.
// Heavy GEMMs: v_wmma_f32_16x16x32_bf16, 4-wave blocks, 64x32 C tiles,
// B slab staged in LDS (double-buffered) via global_load_async_to_lds_b128
// (NT path) or transpose-store (NN path). Non-temporal hints on the 134MB
// logits stream (exceeds the 192MB L2 together with the softmax output).
// ---------------------------------------------------------------------------

typedef __attribute__((ext_vector_type(16))) __bf16 v16bf;
typedef __attribute__((ext_vector_type(8)))  float  v8f;

__device__ __forceinline__ unsigned short f2bf_u16(float f) {
  union { float f; unsigned u; } v; v.f = f;
  unsigned r = v.u + 0x7FFFu + ((v.u >> 16) & 1u);   // round-to-nearest-even
  return (unsigned short)(r >> 16);
}
__device__ __forceinline__ __bf16 u16_bf(unsigned short h) {
  return __builtin_bit_cast(__bf16, h);
}

// ---- WMMA fragment loaders (per CDNA5 ISA VGPR striping) --------
// A: 16x32 bf16 tile from row-major global. lane m=lane&15, half=lane>>4,
// VGPR j holds K pair at kb = (j<4?0:16) + half*8 + (j&3)*2.
__device__ __forceinline__ v16bf frag_a(const unsigned short* base, int lda,
                                        int row0, int col0) {
  int lane = threadIdx.x & 31;
  int m = lane & 15, half = lane >> 4;
  const unsigned short* p = base + (long long)(row0 + m) * lda + col0;
  v16bf f;
#pragma unroll
  for (int j = 0; j < 8; ++j) {
    int kb = ((j & 4) ? 16 : 0) + half * 8 + (j & 3) * 2;
    unsigned pk = *(const unsigned*)(p + kb);
    f[2 * j]     = u16_bf((unsigned short)(pk & 0xffffu));
    f[2 * j + 1] = u16_bf((unsigned short)(pk >> 16));
  }
  return f;
}
// B fragment from the LDS slab, stored as [n][LDSW] with elements [n][k].
// lane n=lane&15, koff=(lane>>4)*8; VGPR j holds K pair at (j<4?0:16) + koff + (j&3)*2.
#define LDSW 40   // 80B rows: 16B-aligned for b128 async writes, bank skew
__device__ __forceinline__ v16bf frag_b_lds(const unsigned short* slab,
                                            int nLocal0) {
  int lane = threadIdx.x & 31;
  int n = (lane & 15) + nLocal0;
  int koff = (lane >> 4) * 8;
  const unsigned short* p = slab + n * LDSW + koff;
  v16bf f;
#pragma unroll
  for (int j = 0; j < 8; ++j) {
    unsigned pk = *(const unsigned*)(p + ((j & 4) ? 16 : 0) + (j & 3) * 2);
    f[2 * j]     = u16_bf((unsigned short)(pk & 0xffffu));
    f[2 * j + 1] = u16_bf((unsigned short)(pk >> 16));
  }
  return f;
}

// ---- Batched WMMA GEMM: C = alpha*A*B(+bias)(+alignMask)(relu) -------------
// Block = 128 threads (4 waves); block tile 64x32 (wave w owns rows +16w).
// B slab (32Kx32N as [n][k]) double-buffered in LDS, shared by all 4 waves.
template <bool TRANSB>
__global__ __launch_bounds__(128)
void gemm_bf16_wmma(const unsigned short* __restrict__ A, int lda, long long sA,
                    const unsigned short* __restrict__ Bm, int ldb, long long sB,
                    float* __restrict__ Cf, unsigned short* __restrict__ Cbf,
                    int ldc, long long sC,
                    const float* __restrict__ bias,
                    const int* __restrict__ amRow, const int* __restrict__ amCol,
                    long long sAM,
                    int M, int N, int K, float alpha, int relu, int ntc) {
  (void)M; (void)N;
  __shared__ unsigned short ldsB[2][32 * LDSW];
  int bz = blockIdx.z;
  A  += (long long)bz * sA;
  Bm += (long long)bz * sB;
  int tid  = threadIdx.x;
  int wave = tid >> 5;
  int m0 = blockIdx.y * 64 + wave * 16;
  int n0 = blockIdx.x * 32;

  // Stage B[k0:k0+32, n0:n0+32] into ldsB[buf] as [n][k].
  auto stage = [&](int buf, int k0) {
    if (TRANSB) {
      // B stored [N][K]: straight 16B copies -> async DMA into LDS.
      int nn = tid >> 2, koff = (tid & 3) * 8;
      const unsigned short* g = Bm + (long long)(n0 + nn) * ldb + k0 + koff;
      unsigned ldsa = (unsigned)(unsigned long long)&ldsB[buf][nn * LDSW + koff];
      asm volatile("global_load_async_to_lds_b128 %0, %1, off"
                   :: "v"(ldsa), "v"(g) : "memory");
    } else {
      // B stored [K][N]: transpose while staging (coalesced global reads).
      int kk = tid >> 2, noff = (tid & 3) * 8;
      const unsigned short* g = Bm + (long long)(k0 + kk) * ldb + n0 + noff;
#pragma unroll
      for (int e = 0; e < 8; ++e)
        ldsB[buf][(noff + e) * LDSW + kk] = g[e];
    }
  };

  v8f acc0 = {0.f, 0.f, 0.f, 0.f, 0.f, 0.f, 0.f, 0.f};
  v8f acc1 = {0.f, 0.f, 0.f, 0.f, 0.f, 0.f, 0.f, 0.f};
  int nk = K / 32;
  stage(0, 0);
  for (int ki = 0; ki < nk; ++ki) {
    int buf = ki & 1;
    if (TRANSB) asm volatile("s_wait_asynccnt 0" ::: "memory");
    __syncthreads();
    if (ki + 1 < nk) stage(buf ^ 1, (ki + 1) * 32);   // prefetch next slab
    v16bf af = frag_a(A, lda, m0, ki * 32);
    v16bf b0 = frag_b_lds(&ldsB[buf][0], 0);
    v16bf b1 = frag_b_lds(&ldsB[buf][0], 16);
    acc0 = __builtin_amdgcn_wmma_f32_16x16x32_bf16(false, af, false, b0,
                                                   (short)0, acc0, false, false);
    acc1 = __builtin_amdgcn_wmma_f32_16x16x32_bf16(false, af, false, b1,
                                                   (short)0, acc1, false, false);
  }

  // C layout: lane n=lane&15, VGPR r holds M = r + 8*(lane>>4).
  int lane = tid & 31;
  int n = lane & 15, mh = lane >> 4;
  const int* amR = amRow ? amRow + (long long)bz * sAM : nullptr;
  const int* amC = amCol ? amCol + (long long)bz * sAM : nullptr;
  long long cbase = (long long)bz * sC;
#pragma unroll
  for (int r = 0; r < 8; ++r) {
    int m = m0 + r + 8 * mh;
#pragma unroll
    for (int t = 0; t < 2; ++t) {
      int nn = n0 + n + 16 * t;
      float v = (t ? acc1[r] : acc0[r]) * alpha;
      if (bias) v += bias[nn];
      if (amR)  v += (amR[m] == amC[nn]) ? 0.0f : -1.0e6f;
      if (relu) v = fmaxf(v, 0.0f);
      long long off = cbase + (long long)m * ldc + nn;
      if (Cf) {
        if (ntc) __builtin_nontemporal_store(v, &Cf[off]);
        else Cf[off] = v;
      }
      if (Cbf) Cbf[off] = f2bf_u16(v);
    }
  }
}

// ---- Elementwise / reduction helpers ---------------------------------------
__global__ void k_f32_to_bf16(const float* __restrict__ in,
                              unsigned short* __restrict__ out, long long n) {
  long long i = (long long)blockIdx.x * blockDim.x + threadIdx.x;
  if (i < n) out[i] = f2bf_u16(in[i]);
}

// out[b][c][r] = bf16(in[b][r][c])  (in: [batch][R][Ccols] fp32)
__global__ void k_transpose_bf16(const float* __restrict__ in,
                                 unsigned short* __restrict__ out,
                                 int R, int Ccols, int batch) {
  long long total = (long long)batch * R * Ccols;
  long long idx = (long long)blockIdx.x * blockDim.x + threadIdx.x;
  if (idx >= total) return;
  int c = (int)(idx % Ccols);
  long long t = idx / Ccols;
  int r = (int)(t % R);
  int b = (int)(t / R);
  out[((long long)b * Ccols + c) * R + r] = f2bf_u16(in[idx]);
}

// out[b][col] = argmax over rows a of s[b][a][col]  (first-max tie break)
__global__ void k_argmax_cols(const float* __restrict__ s, int* __restrict__ out,
                              int Arows, int Ncols, int batch) {
  long long idx = (long long)blockIdx.x * blockDim.x + threadIdx.x;
  if (idx >= (long long)batch * Ncols) return;
  int col = (int)(idx % Ncols);
  int b   = (int)(idx / Ncols);
  const float* p = s + (long long)b * Arows * Ncols + col;
  float best = p[0]; int bi = 0;
  for (int a = 1; a < Arows; ++a) {
    float v = p[(long long)a * Ncols];
    if (v > best) { best = v; bi = a; }
  }
  out[idx] = bi;
}

// out[b][row] = argmax over cols a of s[b][row][a]
__global__ void k_argmax_rows(const float* __restrict__ s, int* __restrict__ out,
                              int Nrows, int Acols, int batch) {
  long long idx = (long long)blockIdx.x * blockDim.x + threadIdx.x;
  if (idx >= (long long)batch * Nrows) return;
  const float* p = s + idx * (long long)Acols;
  float best = p[0]; int bi = 0;
  for (int a = 1; a < Acols; ++a) {
    float v = p[a];
    if (v > best) { best = v; bi = a; }
  }
  out[idx] = bi;
}

// softmax over batch axis (size 2, legacy torch dim=0); result written
// TRANSPOSED bf16: qst[b][j][i] = softmax_b(logits[b][i][j]). NT hints:
// this 200MB stream must not evict the resident GEMM operands from L2.
__global__ void k_softmax_batch2_T(const float* __restrict__ logits,
                                   unsigned short* __restrict__ qst, int HWn) {
  long long total = (long long)HWn * HWn;
  long long idx = (long long)blockIdx.x * blockDim.x + threadIdx.x;
  if (idx >= total) return;
  int i = (int)(idx / HWn), j = (int)(idx % HWn);
  float l0 = __builtin_nontemporal_load(&logits[idx]);
  float l1 = __builtin_nontemporal_load(&logits[total + idx]);
  float mx = fmaxf(l0, l1);
  float e0 = __expf(l0 - mx), e1 = __expf(l1 - mx);
  float inv = 1.0f / (e0 + e1);
  long long o = (long long)j * HWn + i;
  __builtin_nontemporal_store(f2bf_u16(e0 * inv), &qst[o]);
  __builtin_nontemporal_store(f2bf_u16(e1 * inv), &qst[total + o]);
}

// Direct 3x3 SAME conv, NCHW, no bias, optional relu. Tiny vs the GEMMs.
__global__ void k_conv3x3(const float* __restrict__ in, const float* __restrict__ w,
                          float* __restrict__ out, int B, int Cin, int Cout,
                          int Hh, int Ww, int relu) {
  long long total = (long long)B * Cout * Hh * Ww;
  long long idx = (long long)blockIdx.x * blockDim.x + threadIdx.x;
  if (idx >= total) return;
  int x = (int)(idx % Ww); long long t = idx / Ww;
  int y = (int)(t % Hh);   t /= Hh;
  int oc = (int)(t % Cout);
  int b  = (int)(t / Cout);
  const float* inb = in + (long long)b * Cin * Hh * Ww;
  float s = 0.0f;
  for (int ic = 0; ic < Cin; ++ic) {
    const float* ip = inb + (long long)ic * Hh * Ww;
    const float* wp = w + ((long long)oc * Cin + ic) * 9;
#pragma unroll
    for (int kh = 0; kh < 3; ++kh) {
      int yy = y + kh - 1;
      if (yy < 0 || yy >= Hh) continue;
#pragma unroll
      for (int kw = 0; kw < 3; ++kw) {
        int xx = x + kw - 1;
        if (xx < 0 || xx >= Ww) continue;
        s += ip[yy * Ww + xx] * wp[kh * 3 + kw];
      }
    }
  }
  if (relu) s = fmaxf(s, 0.0f);
  out[idx] = s;
}

// ---------------------------------------------------------------------------
static inline int cdiv_ll(long long a, long long b) { return (int)((a + b - 1) / b); }

extern "C" void kernel_launch(void* const* d_in, const int* in_sizes, int n_in,
                              void* d_out, int out_size, void* d_ws, size_t ws_size,
                              hipStream_t stream) {
  (void)in_sizes; (void)n_in; (void)out_size; (void)ws_size;
  const float* tok   = (const float*)d_in[0];
  const float* supp  = (const float*)d_in[1];
  const float* query = (const float*)d_in[2];
  const float* Wqa = (const float*)d_in[3];  const float* bqa = (const float*)d_in[4];
  const float* Wks = (const float*)d_in[5];  const float* bks = (const float*)d_in[6];
  const float* Wka = (const float*)d_in[7];  const float* bka = (const float*)d_in[8];
  const float* Wvs = (const float*)d_in[9];  const float* bvs = (const float*)d_in[10];
  const float* W1  = (const float*)d_in[11]; const float* b1  = (const float*)d_in[12];
  const float* W2  = (const float*)d_in[13]; const float* b2  = (const float*)d_in[14];
  const float* cw3 = (const float*)d_in[15];
  const float* cw1 = (const float*)d_in[16];

  constexpr int B = 2, NA = 128, HW = 4096, C = 256, DFF = 2048, Hh = 64;
  const float scale = 0.1767766952966369f;  // 1/sqrt(32)

  // bump allocator over d_ws (needs ~290 MB)
  char* wp_ = (char*)d_ws;
  auto alloc = [&](size_t bytes) -> void* {
    void* r = (void*)wp_;
    wp_ += (bytes + 255) & ~((size_t)255);
    return r;
  };
  auto aus = [&](long long n) { return (unsigned short*)alloc((size_t)n * 2); };
  auto af  = [&](long long n) { return (float*)alloc((size_t)n * 4); };
  auto ai  = [&](long long n) { return (int*)alloc((size_t)n * 4); };

  unsigned short* tok_h   = aus((long long)B * NA * C);
  unsigned short* supp_h  = aus((long long)B * HW * C);
  unsigned short* query_h = aus((long long)B * HW * C);
  unsigned short* Wqa_h = aus((long long)C * C);
  unsigned short* Wks_h = aus((long long)C * C);
  unsigned short* Wka_h = aus((long long)C * C);
  unsigned short* Wvs_h = aus((long long)C * C);
  unsigned short* W1_h  = aus((long long)C * DFF);
  unsigned short* W2_h  = aus((long long)DFF * C);
  unsigned short* qa_h  = aus((long long)B * NA * C);
  unsigned short* ks_h  = aus((long long)B * HW * C);
  unsigned short* qq_h  = aus((long long)B * HW * C);
  unsigned short* ka_h  = aus((long long)B * NA * C);
  unsigned short* vs_h  = aus((long long)B * HW * C);
  unsigned short* sasT_h = aus((long long)B * HW * NA);  // scores_as^T bf16
  unsigned short* sqa_h  = aus((long long)B * HW * NA);  // scores_qa bf16
  unsigned short* qst_h  = aus((long long)B * HW * HW);  // softmax, transposed
  unsigned short* dec_h  = aus((long long)B * HW * C);
  unsigned short* hid_h  = aus((long long)B * HW * DFF);
  float* sas_f   = af((long long)B * NA * HW);
  float* sqa_f   = af((long long)B * HW * NA);
  float* logits  = af((long long)B * HW * HW);
  float* decfin  = af((long long)B * HW * C);
  float* conv_t1 = af((long long)B * 32 * Hh * Hh);
  int* Aidx = ai((long long)B * HW);
  int* Qidx = ai((long long)B * HW);

  auto cvt = [&](const float* in, unsigned short* out, long long n) {
    k_f32_to_bf16<<<cdiv_ll(n, 256), 256, 0, stream>>>(in, out, n);
  };
  auto gemm = [&](bool transB,
                  const unsigned short* A, int lda, long long sA,
                  const unsigned short* Bm, int ldb, long long sB,
                  float* Cf, unsigned short* Cbf, int ldc, long long sC,
                  const float* bias, const int* amR, const int* amC, long long sAM,
                  int M, int N, int K, float alpha, int relu, int ntc, int batch) {
    dim3 g((unsigned)(N / 32), (unsigned)(M / 64), (unsigned)batch);
    if (transB)
      gemm_bf16_wmma<true><<<g, 128, 0, stream>>>(A, lda, sA, Bm, ldb, sB, Cf, Cbf,
                                                  ldc, sC, bias, amR, amC, sAM,
                                                  M, N, K, alpha, relu, ntc);
    else
      gemm_bf16_wmma<false><<<g, 128, 0, stream>>>(A, lda, sA, Bm, ldb, sB, Cf, Cbf,
                                                   ldc, sC, bias, amR, amC, sAM,
                                                   M, N, K, alpha, relu, ntc);
  };

  // 1) fp32 -> bf16 operand copies
  cvt(tok, tok_h, (long long)B * NA * C);
  cvt(supp, supp_h, (long long)B * HW * C);
  cvt(query, query_h, (long long)B * HW * C);
  cvt(Wqa, Wqa_h, (long long)C * C);
  cvt(Wks, Wks_h, (long long)C * C);
  cvt(Wka, Wka_h, (long long)C * C);
  cvt(Wvs, Wvs_h, (long long)C * C);
  cvt(W1, W1_h, (long long)C * DFF);
  cvt(W2, W2_h, (long long)DFF * C);

  // 2) projections (batch flattened into M; weights shared) -> bf16 outputs
  gemm(false, tok_h,   C, 0, Wqa_h, C, 0, nullptr, qa_h, C, 0, bqa,
       nullptr, nullptr, 0, B * NA, C, C, 1.0f, 0, 0, 1);
  gemm(false, supp_h,  C, 0, Wks_h, C, 0, nullptr, ks_h, C, 0, bks,
       nullptr, nullptr, 0, B * HW, C, C, 1.0f, 0, 0, 1);
  gemm(false, query_h, C, 0, Wqa_h, C, 0, nullptr, qq_h, C, 0, bqa,
       nullptr, nullptr, 0, B * HW, C, C, 1.0f, 0, 0, 1);  // qq reuses Wqa/bqa
  gemm(false, tok_h,   C, 0, Wka_h, C, 0, nullptr, ka_h, C, 0, bka,
       nullptr, nullptr, 0, B * NA, C, C, 1.0f, 0, 0, 1);
  gemm(false, supp_h,  C, 0, Wvs_h, C, 0, nullptr, vs_h, C, 0, bvs,
       nullptr, nullptr, 0, B * HW, C, C, 1.0f, 0, 0, 1);

  // 3) attention scores (NT: both operands row-major over channel dim)
  gemm(true, qa_h, C, (long long)NA * C, ks_h, C, (long long)HW * C,
       sas_f, nullptr, HW, (long long)NA * HW, nullptr, nullptr, nullptr, 0,
       NA, HW, C, scale, 0, 0, B);                              // [B,128,4096]
  gemm(true, qq_h, C, (long long)HW * C, ka_h, C, (long long)NA * C,
       sqa_f, sqa_h, NA, (long long)HW * NA, nullptr, nullptr, nullptr, 0,
       HW, NA, C, scale, 0, 0, B);                              // [B,4096,128]

  // 4) argmaxes + transposed bf16 copy of scores_as
  k_argmax_cols<<<cdiv_ll((long long)B * HW, 256), 256, 0, stream>>>(
      sas_f, Aidx, NA, HW, B);
  k_argmax_rows<<<cdiv_ll((long long)B * HW, 256), 256, 0, stream>>>(
      sqa_f, Qidx, HW, NA, B);
  k_transpose_bf16<<<cdiv_ll((long long)B * NA * HW, 256), 256, 0, stream>>>(
      sas_f, sasT_h, NA, HW, B);

  // 5) logits[b,i,j] = sum_a sas[b,a,i]*sqa[b,j,a] + alignMask (NT, K=128, NT-stores)
  gemm(true, sasT_h, NA, (long long)HW * NA, sqa_h, NA, (long long)HW * NA,
       logits, nullptr, HW, (long long)HW * HW, nullptr, Aidx, Qidx, HW,
       HW, HW, NA, 1.0f, 0, 1, B);

  // 6) legacy softmax over batch axis (dim=0), output transposed bf16
  k_softmax_batch2_T<<<cdiv_ll((long long)HW * HW, 256), 256, 0, stream>>>(
      logits, qst_h, HW);

  // 7) dec[b,j,c] = sum_i qs[b,i,j]*vs[b,i,c]  => (qs^T)@vs, NN, K=4096
  gemm(false, qst_h, HW, (long long)HW * HW, vs_h, C, (long long)HW * C,
       nullptr, dec_h, C, (long long)HW * C, nullptr, nullptr, nullptr, 0,
       HW, C, HW, 1.0f, 0, 0, B);

  // 8) FFN: relu(dec@W1+b1)@W2+b2   (batch flattened into M)
  gemm(false, dec_h, C, 0, W1_h, DFF, 0, nullptr, hid_h, DFF, 0, b1,
       nullptr, nullptr, 0, B * HW, DFF, C, 1.0f, 1, 0, 1);
  gemm(false, hid_h, DFF, 0, W2_h, C, 0, decfin, nullptr, C, 0, b2,
       nullptr, nullptr, 0, B * HW, C, DFF, 1.0f, 0, 0, 1);

  // 9) raw view [B,HW,C] -> [B,C,64,64] is a flat reinterpretation; convs.
  k_conv3x3<<<cdiv_ll((long long)B * 32 * Hh * Hh, 128), 128, 0, stream>>>(
      decfin, cw3, conv_t1, B, C, 32, Hh, Hh, 1);
  k_conv3x3<<<cdiv_ll((long long)B * 3 * Hh * Hh, 128), 128, 0, stream>>>(
      conv_t1, cw1, (float*)d_out, B, 32, 3, Hh, Hh, 0);
}
